// FlowLayer_79336635892634
// MI455X (gfx1250) — matrix operand order repaired and checked by reading, not compiled, not run on hardware
//
#include <hip/hip_runtime.h>
#include <math.h>

// ---------------------------------------------------------------------------
// FlowLayer on gfx1250 (MI455X, wave32, WMMA).
//  N=8, C=256, H=W=16, num=256, EQN=10, GR=4, nb=32, 4*num=1024.
// Pipeline:
//   1) im2col(x) -> bf16; conv1 (att & grad) via v_wmma_f32_16x16x32_bf16
//   2) im2col(hidden) -> conv2 (att sigmoid / grad linear)
//   3) sparse extraction of 'a' (<=2 nnz/row) -> scatter-assemble ATA/ATB
//   4) blocked Cholesky (16x16 blocks), SYRK updates via v_wmma_f32_16x16x4_f32
//   5) fwd/back substitution, GroupNorm + SE scaling, 16->128 post conv
// ---------------------------------------------------------------------------

typedef __attribute__((ext_vector_type(16))) __bf16 v16bf;
typedef __attribute__((ext_vector_type(8)))  __bf16 v8bf;
typedef __attribute__((ext_vector_type(8)))  float  v8f;
typedef __attribute__((ext_vector_type(2)))  float  v2f;

#define NBATCH 32
#define MROWS  2048      // N*H*W
#define KDIM   2304      // 256*9
#define NDIM4  1024      // 4*num

struct RowNZ { int j0, j1; float v0, v1; };

// ---------------- workspace layout (bytes, all 256B aligned) ---------------
constexpr size_t SZ_ATA  = (size_t)NBATCH * 1024 * 1024 * sizeof(float); // 128 MiB
constexpr size_t SZ_VEC  = (size_t)NBATCH * 1024 * sizeof(float);
constexpr size_t OFF_ATA  = 0;
constexpr size_t OFF_ATB  = OFF_ATA + SZ_ATA;
constexpr size_t OFF_Y    = OFF_ATB + SZ_VEC;
constexpr size_t OFF_SOL  = OFF_Y   + SZ_VEC;
constexpr size_t OFF_XO   = OFF_SOL + SZ_VEC;                        // 8*16*256 f32
constexpr size_t OFF_HIDA = OFF_XO   + (size_t)8*16*256*4;
constexpr size_t OFF_HIDG = OFF_HIDA + (size_t)MROWS*256*4;
constexpr size_t OFF_ATT  = OFF_HIDG + (size_t)MROWS*256*4;
constexpr size_t OFF_GRD  = OFF_ATT  + (size_t)MROWS*48*4;
constexpr size_t OFF_IM   = OFF_GRD  + (size_t)MROWS*48*4;
constexpr size_t OFF_W1A  = OFF_IM   + (size_t)MROWS*KDIM*2;
constexpr size_t OFF_W2A  = OFF_W1A  + (size_t)256*KDIM*2;
constexpr size_t OFF_W1G  = OFF_W2A  + (size_t)48*KDIM*2;
constexpr size_t OFF_W2G  = OFF_W1G  + (size_t)256*KDIM*2;
constexpr size_t OFF_POOL = OFF_W2G  + (size_t)48*KDIM*2;
constexpr size_t OFF_SE   = OFF_POOL + (size_t)8*256*4;
constexpr size_t OFF_ROWS = OFF_SE   + (size_t)8*16*4;

// ---------------------------------------------------------------------------
__global__ void fl_zero_f32(float* __restrict__ p, long count) {
  long i = (long)blockIdx.x * blockDim.x + threadIdx.x;
  long s = (long)gridDim.x * blockDim.x;
  for (; i < count; i += s) p[i] = 0.0f;
}

// OIHW f32 -> row-major B^T[Npad][K] bf16  (k = ci*9+kh*3+kw matches flat OIHW)
__global__ void fl_repack_w(const float* __restrict__ w, __bf16* __restrict__ bt,
                            int O, int Npad, int K) {
  long idx = (long)blockIdx.x * blockDim.x + threadIdx.x;
  if (idx >= (long)Npad * K) return;
  int o = (int)(idx / K);
  float v = (o < O) ? w[idx] : 0.0f;
  bt[idx] = (__bf16)v;
}

// im2col -> bf16 patches [2048][2304]; mode 0: NCHW f32 src, mode 1: NHWC f32
__global__ void fl_im2col(const float* __restrict__ src, __bf16* __restrict__ dst,
                          int mode) {
  long idx = (long)blockIdx.x * blockDim.x + threadIdx.x;
  if (idx >= (long)MROWS * KDIM) return;
  int m = (int)(idx / KDIM), k = (int)(idx % KDIM);
  int n = m >> 8, pix = m & 255, y = pix >> 4, x = pix & 15;
  int ci = k / 9, r = k % 9, kh = r / 3, kw = r % 3;
  int yy = y + kh - 1, xx = x + kw - 1;
  float v = 0.0f;
  if (yy >= 0 && yy < 16 && xx >= 0 && xx < 16) {
    if (mode == 0) v = src[(((size_t)n * 256 + ci) * 16 + yy) * 16 + xx];
    else           v = src[(((size_t)n * 16 + yy) * 16 + xx) * 256 + ci];
  }
  dst[idx] = (__bf16)v;
}

// C[M][Npad] = act(A[M][K] x B^T[Npad][K]^T + bias); one wave per 16x16 tile.
// act: 0=none, 1=leaky_relu(0.01), 2=sigmoid
__global__ __launch_bounds__(32)
void fl_gemm_bf16(const __bf16* __restrict__ A, const __bf16* __restrict__ BT,
                  const float* __restrict__ bias, float* __restrict__ C,
                  int K, int Npad, int biasN, int act) {
  const int lane = threadIdx.x;
  const int rsel = lane & 15;
  const int hf   = lane >> 4;
  const long m0 = (long)blockIdx.y * 16;
  const long n0 = (long)blockIdx.x * 16;
  const __bf16* arow = A  + (m0 + rsel) * (long)K + hf * 16;
  const __bf16* brow = BT + (n0 + rsel) * (long)K + hf * 16;
  v8f acc = {};
  for (int k0 = 0; k0 < K; k0 += 32) {
    if ((k0 & 255) == 0) __builtin_prefetch(arow + k0 + 512, 0, 0);
    // A 16x32 bf16 fragment: per-lane K contiguous {k0+hf*16 .. +16}
    v8bf alo = *(const v8bf*)(arow + k0);
    v8bf ahi = *(const v8bf*)(arow + k0 + 8);
    v16bf af;
#pragma unroll
    for (int i = 0; i < 8; ++i) { af[i] = alo[i]; af[i + 8] = ahi[i]; }
    // B 32x16 bf16 fragment: lane holds column n, K contiguous (via B^T)
    v16bf bfr = *(const v16bf*)(brow + k0);
    acc = __builtin_amdgcn_wmma_f32_16x16x32_bf16(false, af, false, bfr,
                                                  (short)0, acc, false, false);
  }
  const int n = (int)n0 + rsel;
  const float bv = (n < biasN) ? bias[n] : 0.0f;
#pragma unroll
  for (int v = 0; v < 8; ++v) {
    float xv = acc[v] + bv;
    if (act == 1)      xv = xv > 0.0f ? xv : 0.01f * xv;
    else if (act == 2) xv = 1.0f / (1.0f + __expf(-xv));
    C[(m0 + hf * 8 + v) * (long)Npad + n] = xv;
  }
}

// Extract <=2 nonzeros per row of the dense FD operator a[1,256,10,1024].
__global__ void fl_rows_extract(const float* __restrict__ a, RowNZ* __restrict__ rows) {
  int r = blockIdx.x * blockDim.x + threadIdx.x;
  if (r >= 2560) return;
  const float* ar = a + (size_t)r * 1024;
  int j0 = -1, j1 = -1; float v0 = 0.0f, v1 = 0.0f;
  for (int j = 0; j < 1024; ++j) {
    float v = ar[j];
    if (v != 0.0f) {
      if (j0 < 0)      { j0 = j; v0 = v; }
      else if (j1 < 0) { j1 = j; v1 = v; }
    }
  }
  rows[r] = {j0, j1, v0, v1};
}

// Scatter-assemble ATA/ATB exploiting 2:row sparsity. One block per batch.
__global__ void fl_assemble(const RowNZ* __restrict__ rows,
                            const float* __restrict__ ATT, const float* __restrict__ GRD,
                            float* __restrict__ ATA, float* __restrict__ ATB) {
  const int b = blockIdx.x, n = b >> 2, g = b & 3;
  float* Ab = ATA + (size_t)b * 1024 * 1024;
  float* Bb = ATB + (size_t)b * 1024;
  for (int r = threadIdx.x; r < 2560; r += blockDim.x) {
    RowNZ rn = rows[r];
    if (rn.j0 < 0) continue;
    int i = r / 10, e = r % 10, c = g * 10 + e;
    size_t off = (size_t)(n * 256 + i) * 48 + c;
    float w  = ATT[off];
    float gr = GRD[off];
    float w2 = w * w;
    float bsc = w2 * gr;
    atomicAdd(&Ab[(size_t)rn.j0 * 1024 + rn.j0], w2 * rn.v0 * rn.v0);
    atomicAdd(&Bb[rn.j0], bsc * rn.v0);
    if (rn.j1 >= 0) {
      float cr = w2 * rn.v0 * rn.v1;
      atomicAdd(&Ab[(size_t)rn.j1 * 1024 + rn.j1], w2 * rn.v1 * rn.v1);
      atomicAdd(&Ab[(size_t)rn.j0 * 1024 + rn.j1], cr);
      atomicAdd(&Ab[(size_t)rn.j1 * 1024 + rn.j0], cr);
      atomicAdd(&Bb[rn.j1], bsc * rn.v1);
    }
  }
}

__global__ void fl_jitter(float* __restrict__ ATA) {
  int idx = blockIdx.x * blockDim.x + threadIdx.x;
  if (idx >= NBATCH * 1024) return;
  int b = idx >> 10, d = idx & 1023;
  ATA[(size_t)b * 1048576 + (size_t)d * 1024 + d] += 1e-12f;
}

// ----- blocked Cholesky: 16x16 blocks, lower triangle stored in-place ------
__global__ __launch_bounds__(32)
void fl_chol_diag(float* __restrict__ ATA, int kb) {
  float* D = ATA + (size_t)blockIdx.x * 1048576 + (size_t)(kb * 16) * 1024 + kb * 16;
  const int lane = threadIdx.x;
  __shared__ float t_[16][17];
#pragma unroll
  for (int v = 0; v < 8; ++v) {
    int e = lane + v * 32;
    t_[e >> 4][e & 15] = D[(size_t)(e >> 4) * 1024 + (e & 15)];
  }
  __syncthreads();
  for (int c = 0; c < 16; ++c) {
    float d = sqrtf(fmaxf(t_[c][c], 1e-30f));
    __syncthreads();
    if (lane < 16) {
      if (lane == c)     t_[c][c] = d;
      else if (lane > c) t_[lane][c] /= d;
    }
    __syncthreads();
    int cnt = (15 - c) * (16 - c) / 2;
    for (int e = lane; e < cnt; e += 32) {
      int rr = 0; while ((rr + 1) * (rr + 2) / 2 <= e) ++rr;
      int cc = e - rr * (rr + 1) / 2;
      int R = c + 1 + rr, Cc = c + 1 + cc;
      t_[R][Cc] -= t_[R][c] * t_[Cc][c];
    }
    __syncthreads();
  }
#pragma unroll
  for (int v = 0; v < 8; ++v) {
    int e = lane + v * 32, r = e >> 4, c = e & 15;
    if (c <= r) D[(size_t)r * 1024 + c] = t_[r][c];
  }
}

// panel: B <- B * L11^{-T}; each lane owns one row of the 16x16 block.
__global__ __launch_bounds__(32)
void fl_chol_trsm(float* __restrict__ ATA, int kb) {
  float* Ab = ATA + (size_t)blockIdx.y * 1048576;
  const int bi = kb + 1 + blockIdx.x;
  const float* L11 = Ab + (size_t)(kb * 16) * 1024 + kb * 16;
  float* B = Ab + (size_t)(bi * 16) * 1024 + kb * 16;
  const int lane = threadIdx.x;
  __shared__ float l_[16][17];
#pragma unroll
  for (int v = 0; v < 8; ++v) {
    int e = lane + v * 32;
    l_[e >> 4][e & 15] = L11[(size_t)(e >> 4) * 1024 + (e & 15)];
  }
  __syncthreads();
  if (lane < 16) {
    float xr[16];
    float* brow = B + (size_t)lane * 1024;
#pragma unroll
    for (int c = 0; c < 16; ++c) xr[c] = brow[c];
#pragma unroll
    for (int c = 0; c < 16; ++c) {
      float s = xr[c];
      for (int k = 0; k < c; ++k) s -= xr[k] * l_[c][k];
      xr[c] = s / l_[c][c];
    }
#pragma unroll
    for (int c = 0; c < 16; ++c) brow[c] = xr[c];
  }
}

// trailing update C(bi,bj) -= L(bi,kb) * L(bj,kb)^T via fp32 WMMA 16x16x4.
// fp32 WMMA has no A/B negation (NEG = {CNeg,0,0}); accumulate P = Li*Lj^T
// from zero, then store C - P.
__global__ __launch_bounds__(32)
void fl_chol_syrk(float* __restrict__ ATA, int kb) {
  float* Ab = ATA + (size_t)blockIdx.y * 1048576;
  int t = blockIdx.x;
  int i = (int)((sqrtf(8.0f * (float)t + 1.0f) - 1.0f) * 0.5f);
  while ((i + 1) * (i + 2) / 2 <= t) ++i;
  while (i * (i + 1) / 2 > t) --i;
  int j = t - i * (i + 1) / 2;
  const int bi = kb + 1 + i, bj = kb + 1 + j;
  const int lane = threadIdx.x;
  const int rc = lane & 15, hf = lane >> 4;
  const float* Li = Ab + (size_t)(bi * 16 + rc) * 1024 + kb * 16;
  const float* Lj = Ab + (size_t)(bj * 16 + rc) * 1024 + kb * 16;
  float* Ct = Ab + (size_t)(bi * 16) * 1024 + bj * 16;
  v8f acc = {};
#pragma unroll
  for (int k0 = 0; k0 < 16; k0 += 4) {
    // A 16x4 f32 fragment: lane row rc, K = k0 + hf*2 + {0,1}
    v2f av = *(const v2f*)(Li + k0 + hf * 2);
    // B 4x16 fragment of Lj^T == A-style fragment of Lj (same pattern)
    v2f bv = *(const v2f*)(Lj + k0 + hf * 2);
    acc = __builtin_amdgcn_wmma_f32_16x16x4_f32(false, av, false, bv,
                                                (short)0, acc, false, false);
  }
#pragma unroll
  for (int v = 0; v < 8; ++v) {
    size_t o = (size_t)(hf * 8 + v) * 1024 + rc;
    Ct[o] = Ct[o] - acc[v];
  }
}

// L y = rhs ; L^T s = y   (one block / batch, parallel dot + LDS reduction)
__global__ void fl_trisolve(const float* __restrict__ ATA, const float* __restrict__ ATB,
                            float* __restrict__ Y, float* __restrict__ SOL) {
  const int b = blockIdx.x, tid = threadIdx.x;
  const float* L = ATA + (size_t)b * 1048576;
  const float* rhs = ATB + (size_t)b * 1024;
  float* y = Y + (size_t)b * 1024;
  float* s = SOL + (size_t)b * 1024;
  __shared__ float red[256];
  for (int i = 0; i < 1024; ++i) {
    float p = 0.0f;
    for (int jj = tid; jj < i; jj += 256) p += L[(size_t)i * 1024 + jj] * y[jj];
    red[tid] = p; __syncthreads();
    for (int o = 128; o > 0; o >>= 1) { if (tid < o) red[tid] += red[tid + o]; __syncthreads(); }
    if (tid == 0) y[i] = (rhs[i] - red[0]) / L[(size_t)i * 1024 + i];
    __syncthreads();
  }
  for (int i = 1023; i >= 0; --i) {
    float p = 0.0f;
    for (int jj = i + 1 + tid; jj < 1024; jj += 256) p += L[(size_t)jj * 1024 + i] * s[jj];
    red[tid] = p; __syncthreads();
    for (int o = 128; o > 0; o >>= 1) { if (tid < o) red[tid] += red[tid + o]; __syncthreads(); }
    if (tid == 0) s[i] = (y[i] - red[0]) / L[(size_t)i * 1024 + i];
    __syncthreads();
  }
}

// ------------------------- SE branch + epilogue ----------------------------
__global__ void fl_pool(const float* __restrict__ x, float* __restrict__ pooled) {
  int n = blockIdx.x, c = threadIdx.x;
  const float* p = x + ((size_t)n * 256 + c) * 256;
  float s = 0.0f;
  for (int i = 0; i < 256; ++i) s += p[i];
  pooled[n * 256 + c] = s * (1.0f / 256.0f);
}

__global__ void fl_se(const float* __restrict__ pooled,
                      const float* __restrict__ w1, const float* __restrict__ b1,
                      const float* __restrict__ w2, const float* __restrict__ b2,
                      float* __restrict__ se) {
  int n = blockIdx.x, tid = threadIdx.x;
  __shared__ float ps[256], h1[128];
  ps[tid] = pooled[n * 256 + tid];
  ps[tid + 128] = pooled[n * 256 + tid + 128];
  __syncthreads();
  float s = b1[tid];
  for (int k = 0; k < 256; ++k) s += w1[tid * 256 + k] * ps[k];
  h1[tid] = s > 0.0f ? s : 0.01f * s;
  __syncthreads();
  if (tid < 16) {
    float t = b2[tid];
    for (int k = 0; k < 128; ++k) t += w2[tid * 128 + k] * h1[k];
    se[n * 16 + tid] = 1.0f / (1.0f + __expf(-t));
  }
}

// sol[32,1024] -> xo[8,16,16,16], GroupNorm(1 group) + affine + SE scale
__global__ void fl_gnorm_se(const float* __restrict__ SOL,
                            const float* __restrict__ gn_w, const float* __restrict__ gn_b,
                            const float* __restrict__ se, float* __restrict__ XO) {
  int n = blockIdx.x, tid = threadIdx.x;
  __shared__ float rs_[256], rq_[256];
  float s = 0.0f, q = 0.0f;
  for (int idx = tid; idx < 4096; idx += 256) {
    int c = idx >> 8, pix = idx & 255;
    float v = SOL[(size_t)(n * 4 + (c >> 2)) * 1024 + (c & 3) * 256 + pix];
    s += v; q += v * v;
  }
  rs_[tid] = s; rq_[tid] = q; __syncthreads();
  for (int o = 128; o > 0; o >>= 1) {
    if (tid < o) { rs_[tid] += rs_[tid + o]; rq_[tid] += rq_[tid + o]; }
    __syncthreads();
  }
  float mu = rs_[0] * (1.0f / 4096.0f);
  float var = rq_[0] * (1.0f / 4096.0f) - mu * mu;
  float inv = rsqrtf(var + 1e-5f);
  for (int idx = tid; idx < 4096; idx += 256) {
    int c = idx >> 8, pix = idx & 255;
    float v = SOL[(size_t)(n * 4 + (c >> 2)) * 1024 + (c & 3) * 256 + pix];
    v = (v - mu) * inv * gn_w[c] + gn_b[c];
    v *= se[n * 16 + c];
    XO[(size_t)(n * 16 + c) * 256 + pix] = v;
  }
}

__global__ void fl_post_conv(const float* __restrict__ XO, const float* __restrict__ w,
                             const float* __restrict__ b, float* __restrict__ out) {
  int idx = blockIdx.x * blockDim.x + threadIdx.x;
  if (idx >= 8 * 128 * 256) return;
  int pix = idx & 255, co = (idx >> 8) & 127, n = idx >> 15;
  int y = pix >> 4, x = pix & 15;
  float acc = b[co];
  for (int ci = 0; ci < 16; ++ci)
    for (int kh = 0; kh < 3; ++kh) {
      int yy = y + kh - 1; if (yy < 0 || yy > 15) continue;
      for (int kw = 0; kw < 3; ++kw) {
        int xx = x + kw - 1; if (xx < 0 || xx > 15) continue;
        acc += XO[(size_t)(n * 16 + ci) * 256 + yy * 16 + xx] *
               w[((co * 16 + ci) * 3 + kh) * 3 + kw];
      }
    }
  out[idx] = acc;
}

// ---------------------------------------------------------------------------
extern "C" void kernel_launch(void* const* d_in, const int* in_sizes, int n_in,
                              void* d_out, int out_size, void* d_ws, size_t ws_size,
                              hipStream_t stream) {
  (void)in_sizes; (void)n_in; (void)out_size; (void)ws_size;
  const float* x       = (const float*)d_in[0];
  const float* a_op    = (const float*)d_in[1];
  const float* att_w1  = (const float*)d_in[2];
  const float* att_b1  = (const float*)d_in[3];
  const float* att_w2  = (const float*)d_in[4];
  const float* att_b2  = (const float*)d_in[5];
  const float* grad_w1 = (const float*)d_in[6];
  const float* grad_b1 = (const float*)d_in[7];
  const float* grad_w2 = (const float*)d_in[8];
  const float* grad_b2 = (const float*)d_in[9];
  const float* se_w1   = (const float*)d_in[10];
  const float* se_b1   = (const float*)d_in[11];
  const float* se_w2   = (const float*)d_in[12];
  const float* se_b2   = (const float*)d_in[13];
  const float* gn_w    = (const float*)d_in[14];
  const float* gn_b    = (const float*)d_in[15];
  const float* post_w  = (const float*)d_in[16];
  const float* post_b  = (const float*)d_in[17];
  float* out = (float*)d_out;
  char*  ws  = (char*)d_ws;

  float*  ATA  = (float*)(ws + OFF_ATA);
  float*  ATB  = (float*)(ws + OFF_ATB);
  float*  Yv   = (float*)(ws + OFF_Y);
  float*  SOL  = (float*)(ws + OFF_SOL);
  float*  XO   = (float*)(ws + OFF_XO);
  float*  HIDA = (float*)(ws + OFF_HIDA);
  float*  HIDG = (float*)(ws + OFF_HIDG);
  float*  ATT  = (float*)(ws + OFF_ATT);
  float*  GRD  = (float*)(ws + OFF_GRD);
  __bf16* IM   = (__bf16*)(ws + OFF_IM);
  __bf16* W1A  = (__bf16*)(ws + OFF_W1A);
  __bf16* W2A  = (__bf16*)(ws + OFF_W2A);
  __bf16* W1G  = (__bf16*)(ws + OFF_W1G);
  __bf16* W2G  = (__bf16*)(ws + OFF_W2G);
  float*  POOL = (float*)(ws + OFF_POOL);
  float*  SE   = (float*)(ws + OFF_SE);
  RowNZ*  ROWS = (RowNZ*)(ws + OFF_ROWS);

  // 0) zero ATA + ATB (contiguous)
  fl_zero_f32<<<8192, 256, 0, stream>>>(ATA, (long)NBATCH * 1024 * 1024 + NBATCH * 1024);

  // 1) weight repack to bf16 B^T
  {
    long n1 = (long)256 * KDIM, n2 = (long)48 * KDIM;
    fl_repack_w<<<(unsigned)((n1 + 255) / 256), 256, 0, stream>>>(att_w1,  W1A, 256, 256, KDIM);
    fl_repack_w<<<(unsigned)((n1 + 255) / 256), 256, 0, stream>>>(grad_w1, W1G, 256, 256, KDIM);
    fl_repack_w<<<(unsigned)((n2 + 255) / 256), 256, 0, stream>>>(att_w2,  W2A, 40, 48, KDIM);
    fl_repack_w<<<(unsigned)((n2 + 255) / 256), 256, 0, stream>>>(grad_w2, W2G, 40, 48, KDIM);
  }

  const unsigned imBlocks = (unsigned)(((long)MROWS * KDIM + 255) / 256);

  // 2) conv1 for both branches (shared im2col of x)
  fl_im2col<<<imBlocks, 256, 0, stream>>>(x, IM, 0);
  fl_gemm_bf16<<<dim3(16, 128), 32, 0, stream>>>(IM, W1A, att_b1,  HIDA, KDIM, 256, 256, 1);
  fl_gemm_bf16<<<dim3(16, 128), 32, 0, stream>>>(IM, W1G, grad_b1, HIDG, KDIM, 256, 256, 1);

  // 3) conv2: att (sigmoid) then grad (linear)
  fl_im2col<<<imBlocks, 256, 0, stream>>>(HIDA, IM, 1);
  fl_gemm_bf16<<<dim3(3, 128), 32, 0, stream>>>(IM, W2A, att_b2,  ATT, KDIM, 48, 40, 2);
  fl_im2col<<<imBlocks, 256, 0, stream>>>(HIDG, IM, 1);
  fl_gemm_bf16<<<dim3(3, 128), 32, 0, stream>>>(IM, W2G, grad_b2, GRD, KDIM, 48, 40, 0);

  // 4) sparse assembly of ATA / ATB
  fl_rows_extract<<<10, 256, 0, stream>>>(a_op, ROWS);
  fl_assemble<<<NBATCH, 256, 0, stream>>>(ROWS, ATT, GRD, ATA, ATB);
  fl_jitter<<<128, 256, 0, stream>>>(ATA);

  // 5) blocked Cholesky over 64 block-columns, 32 batches in parallel
  for (int kb = 0; kb < 64; ++kb) {
    fl_chol_diag<<<NBATCH, 32, 0, stream>>>(ATA, kb);
    int T = 63 - kb;
    if (T > 0) {
      fl_chol_trsm<<<dim3((unsigned)T, NBATCH), 32, 0, stream>>>(ATA, kb);
      int ntiles = T * (T + 1) / 2;
      fl_chol_syrk<<<dim3((unsigned)ntiles, NBATCH), 32, 0, stream>>>(ATA, kb);
    }
  }

  // 6) triangular solves
  fl_trisolve<<<NBATCH, 256, 0, stream>>>(ATA, ATB, Yv, SOL);

  // 7) SE branch
  fl_pool<<<8, 256, 0, stream>>>(x, POOL);
  fl_se<<<8, 128, 0, stream>>>(POOL, se_w1, se_b1, se_w2, se_b2, SE);

  // 8) GroupNorm + SE scale, then post conv 16->128
  fl_gnorm_se<<<8, 256, 0, stream>>>(SOL, gn_w, gn_b, SE, XO);
  fl_post_conv<<<1024, 256, 0, stream>>>(XO, post_w, post_b, out);
}
